// RNNPrediction_3728031613438
// MI455X (gfx1250) — matrix operand order repaired and checked by reading, not compiled, not run on hardware
//
#include <hip/hip_runtime.h>
#include <math.h>

// ---------------------------------------------------------------------------
// MI455X (gfx1250) persistent-kernel RNN:
//   - weights converted once to f16 WMMA A-fragments (104 MB -> fits 192MB L2)
//   - v_wmma_f32_16x16x32_f16 GEMV tiles, fp32 accumulation
//   - one persistent kernel, atomic grid barrier between the 5 phases/step
//   - k-chunk is blockIdx-derived (scalar) so inner loops have no divergence
// ---------------------------------------------------------------------------

typedef __attribute__((ext_vector_type(16))) _Float16 v16h;
typedef __attribute__((ext_vector_type(8)))  float    v8f;

static constexpr int kSteps   = 96;
static constexpr int kH       = 4096;
static constexpr int kI       = 1024;
static constexpr int kNBlk    = 128;   // resident even on small configs (<= 8 WG/WGP @16 WGPs)
static constexpr int kTPB     = 256;   // 8 wave32 per block -> 1024 waves total
static constexpr int kRowT    = kH / 16;        // 256 row tiles
static constexpr int kKT0     = (kI + kH) / 32; // 160 k-tiles, layer 0 (=[W_ih0|W_hh0])
static constexpr int kKT1     = (kH + kH) / 32; // 256 k-tiles, layer 1 (=[W_ih1|W_hh1])
static constexpr int kKSplit  = 4;              // 256 rowtiles * 4 kchunks = 1024 waves
static constexpr int kWinKT   = kI / 32;        // 32 window k-tiles in layer 0

// -------------------- prep: retile f32 weights into f16 WMMA A-fragments ----
// A-fragment as packed here (contiguous k, not AMD's table order): lane<16 row=M=lane holds K {0..15},
// lane>=16 row=lane-16 holds K {16..31}; half h of v16h -> VGPR h/2, slot h%2.
__global__ void __launch_bounds__(kTPB)
tile_weights(const float* __restrict__ src, _Float16* __restrict__ dst,
             int ktiles_src, int Ksrc, int ktiles_layer, int ktile_off, long n)
{
    long idx = (long)blockIdx.x * blockDim.x + threadIdx.x;
    if (idx >= n) return;
    int  s    = (int)(idx & 511);          // slot within 512-half tile
    long tile = idx >> 9;
    int  kt   = (int)(tile % ktiles_src);
    long rt   = tile / ktiles_src;
    int  lane = s >> 4;
    int  h    = s & 15;
    long row  = rt * 16 + (lane & 15);
    int  kloc = h + ((lane >= 16) ? 16 : 0);
    long k    = (long)kt * 32 + kloc;
    long dt   = rt * ktiles_layer + (ktile_off + kt);
    dst[dt * 512 + s] = (_Float16)src[row * (long)Ksrc + k];
}

// -------------------- prep: state init ---------------------------------------
__global__ void __launch_bounds__(kTPB)
init_state(const float* __restrict__ x,
           const float* __restrict__ b_ih0, const float* __restrict__ b_hh0,
           const float* __restrict__ b_ih1, const float* __restrict__ b_hh1,
           float* z0, float* z1, float* bias0, float* bias1,
           _Float16* h01, _Float16* xwin, float* yacc, unsigned* bar)
{
    int i = blockIdx.x * blockDim.x + threadIdx.x;   // 4096 threads
    if (i < kH) {
        z0[i] = 0.0f;  z1[i] = 0.0f;
        bias0[i] = b_ih0[i] + b_hh0[i];
        bias1[i] = b_ih1[i] + b_hh1[i];
        h01[i] = (_Float16)0.0f;          // h0 init
        h01[kH + i] = (_Float16)0.0f;     // h1 init
    }
    if (i < 2048) xwin[i] = (i < kI) ? (_Float16)x[i] : (_Float16)0.0f;
    if (i == 0) { *yacc = 0.0f; bar[0] = 0u; bar[1] = 0u; }
}

// -------------------- grid-wide barrier (device-scope atomics) ---------------
__device__ __forceinline__ void gridBarrier(unsigned* bar)
{
    __syncthreads();
    __threadfence();   // release this block's stores (agent scope)
    if (threadIdx.x == 0) {
        unsigned g = __hip_atomic_load(&bar[1], __ATOMIC_RELAXED, __HIP_MEMORY_SCOPE_AGENT);
        unsigned a = __hip_atomic_fetch_add(&bar[0], 1u, __ATOMIC_ACQ_REL, __HIP_MEMORY_SCOPE_AGENT);
        if (a == (unsigned)kNBlk - 1u) {
            __hip_atomic_store(&bar[0], 0u, __ATOMIC_RELAXED, __HIP_MEMORY_SCOPE_AGENT);
            __hip_atomic_fetch_add(&bar[1], 1u, __ATOMIC_RELEASE, __HIP_MEMORY_SCOPE_AGENT);
        } else {
            while (__hip_atomic_load(&bar[1], __ATOMIC_ACQUIRE, __HIP_MEMORY_SCOPE_AGENT) == g)
                __builtin_amdgcn_s_sleep(2);
        }
    }
    __syncthreads();
    __threadfence();   // acquire remote stores before proceeding
}

// -------------------- persistent RNN kernel ----------------------------------
__global__ void __launch_bounds__(kTPB)
rnn_persistent(const _Float16* __restrict__ wl0, const _Float16* __restrict__ wl1,
               float* z0, float* z1,
               const float* __restrict__ bias0, const float* __restrict__ bias1,
               _Float16* h01, _Float16* xwin, float* yacc, unsigned* bar,
               const float* __restrict__ Wfc, const float* __restrict__ bfc,
               float* __restrict__ out)
{
    __shared__ float red[kTPB];
    const int lane   = threadIdx.x & 31;
    const int wv     = threadIdx.x >> 5;              // wave in block, 0..7
    const int gtid   = blockIdx.x * kTPB + threadIdx.x;
    const int kc     = blockIdx.x >> 5;               // k-chunk 0..3 (block-uniform -> SGPR)
    const int rt     = (blockIdx.x & 31) * 8 + wv;    // row tile 0..255 (wave-uniform)
    const int laneHi = (lane & 16) ? 16 : 0;          // B layout: lanes>=16 hold K 16..31

    // Layer-0 split of this k-chunk into window region [0,32) and h0 region [32,160)
    const int kt0A   = kc * (kKT0 / kKSplit);
    const int kt1A   = kt0A + (kKT0 / kKSplit);
    const int winEnd = (kt1A < kWinKT) ? kt1A : kWinKT;   // window tiles: [kt0A, winEnd)
    const int hBeg   = (kt0A > kWinKT) ? kt0A : kWinKT;   // h0 tiles:     [hBeg, kt1A)
    const int kt0C   = kc * (kKT1 / kKSplit);
    const int kt1C   = kt0C + (kKT1 / kKSplit);

    const _Float16* aBase0 = wl0 + ((long)rt * kKT0 << 9) + (lane << 4);
    const _Float16* aBase1 = wl1 + ((long)rt * kKT1 << 9) + (lane << 4);

    for (int t = 0; t < kSteps; ++t) {
        // ---- Phase A: layer0 partial GEMV: z0 += Wl0[16 rows] * [xwin(t); h0]
        {
            v8f c = {};
            // window region (unaligned base in xwin; uniform trip count)
            for (int kt = kt0A; kt < winEnd; ++kt) {
                v16h a = *(const v16h*)(aBase0 + ((long)kt << 9));
                const _Float16* bs = xwin + (t + kt * 32 + laneHi);
                v16h b;
                #pragma unroll
                for (int h = 0; h < 16; ++h) b[h] = bs[h];
                c = __builtin_amdgcn_wmma_f32_16x16x32_f16(false, a, false, b,
                                                           (short)0, c, false, false);
            }
            // h0 region (32B-aligned vector loads; uniform trip count)
            for (int kt = hBeg; kt < kt1A; ++kt) {
                v16h a = *(const v16h*)(aBase0 + ((long)kt << 9));
                v16h b = *(const v16h*)(h01 + (kt * 32 - kI + laneHi));
                c = __builtin_amdgcn_wmma_f32_16x16x32_f16(false, a, false, b,
                                                           (short)0, c, false, false);
            }
            if ((lane & 15) == 0) {            // column N=0: lane0 -> M0..7, lane16 -> M8..15
                float* zp = z0 + rt * 16 + (laneHi >> 1);
                #pragma unroll
                for (int g = 0; g < 8; ++g) atomicAdd(zp + g, c[g]);
            }
        }
        gridBarrier(bar);

        // ---- Phase B: h0 = tanh(z0 + b_ih0 + b_hh0); reset z0
        if (gtid < kH) {
            float v = tanhf(z0[gtid] + bias0[gtid]);
            h01[gtid] = (_Float16)v;
            z0[gtid] = 0.0f;
        }
        gridBarrier(bar);

        // ---- Phase C: layer1 partial GEMV: z1 += Wl1[16 rows] * [h0; h1]
        {
            v8f c = {};
            for (int kt = kt0C; kt < kt1C; ++kt) {
                v16h a = *(const v16h*)(aBase1 + ((long)kt << 9));
                v16h b = *(const v16h*)(h01 + (kt * 32 + laneHi));
                c = __builtin_amdgcn_wmma_f32_16x16x32_f16(false, a, false, b,
                                                           (short)0, c, false, false);
            }
            if ((lane & 15) == 0) {
                float* zp = z1 + rt * 16 + (laneHi >> 1);
                #pragma unroll
                for (int g = 0; g < 8; ++g) atomicAdd(zp + g, c[g]);
            }
        }
        gridBarrier(bar);

        // ---- Phase D: h1 = tanh(z1 + biases); reset z1; fc partial dot
        {
            float part = 0.0f;
            if (gtid < kH) {
                float v = tanhf(z1[gtid] + bias1[gtid]);
                h01[kH + gtid] = (_Float16)v;
                z1[gtid] = 0.0f;
                part = Wfc[gtid] * v;
            }
            red[threadIdx.x] = part;
            __syncthreads();
            for (int s = kTPB / 2; s > 0; s >>= 1) {
                if (threadIdx.x < s) red[threadIdx.x] += red[threadIdx.x + s];
                __syncthreads();
            }
            if (threadIdx.x == 0) atomicAdd(yacc, red[0]);
        }
        gridBarrier(bar);

        // ---- Phase E: publish y, append to sliding window, reset accumulator
        if (gtid == 0) {
            float y = *yacc + bfc[0];
            out[t] = y;
            xwin[kI + t] = (_Float16)y;
            *yacc = 0.0f;
        }
        gridBarrier(bar);
    }
}

// -------------------- host launch --------------------------------------------
extern "C" void kernel_launch(void* const* d_in, const int* in_sizes, int n_in,
                              void* d_out, int out_size, void* d_ws, size_t ws_size,
                              hipStream_t stream)
{
    (void)in_sizes; (void)n_in; (void)out_size;
    const float* x     = (const float*)d_in[0];
    const float* W_ih0 = (const float*)d_in[1];
    const float* b_ih0 = (const float*)d_in[2];
    const float* W_hh0 = (const float*)d_in[3];
    const float* b_hh0 = (const float*)d_in[4];
    const float* W_ih1 = (const float*)d_in[5];
    const float* b_ih1 = (const float*)d_in[6];
    const float* W_hh1 = (const float*)d_in[7];
    const float* b_hh1 = (const float*)d_in[8];
    const float* W_fc  = (const float*)d_in[9];
    const float* b_fc  = (const float*)d_in[10];
    float* out = (float*)d_out;

    char*  ws  = (char*)d_ws;
    size_t off = 0;
    auto alloc = [&](size_t bytes) {
        size_t o = off; off += (bytes + 255) & ~(size_t)255; return o;
    };
    size_t oWL0 = alloc((size_t)kRowT * kKT0 * 512 * 2);   // 40 MB f16 layer0 tiles
    size_t oWL1 = alloc((size_t)kRowT * kKT1 * 512 * 2);   // 64 MB f16 layer1 tiles
    size_t oZ0  = alloc((size_t)kH * 4);
    size_t oZ1  = alloc((size_t)kH * 4);
    size_t oB0  = alloc((size_t)kH * 4);
    size_t oB1  = alloc((size_t)kH * 4);
    size_t oH01 = alloc((size_t)2 * kH * 2);
    size_t oXW  = alloc((size_t)2048 * 2);
    size_t oY   = alloc(256);
    size_t oBar = alloc(256);
    if (ws_size < off) return;   // workspace too small for f16 weight cache

    _Float16* wl0   = (_Float16*)(ws + oWL0);
    _Float16* wl1   = (_Float16*)(ws + oWL1);
    float*    z0    = (float*)(ws + oZ0);
    float*    z1    = (float*)(ws + oZ1);
    float*    bias0 = (float*)(ws + oB0);
    float*    bias1 = (float*)(ws + oB1);
    _Float16* h01   = (_Float16*)(ws + oH01);
    _Float16* xwin  = (_Float16*)(ws + oXW);
    float*    yacc  = (float*)(ws + oY);
    unsigned* bar   = (unsigned*)(ws + oBar);

    init_state<<<16, kTPB, 0, stream>>>(x, b_ih0, b_hh0, b_ih1, b_hh1,
                                        z0, z1, bias0, bias1, h01, xwin, yacc, bar);

    long n;
    n = (long)kH * kI;  // W_ih0 -> layer0 ktiles [0,32)
    tile_weights<<<(unsigned)((n + kTPB - 1) / kTPB), kTPB, 0, stream>>>(
        W_ih0, wl0, kI / 32, kI, kKT0, 0, n);
    n = (long)kH * kH;  // W_hh0 -> layer0 ktiles [32,160)
    tile_weights<<<(unsigned)((n + kTPB - 1) / kTPB), kTPB, 0, stream>>>(
        W_hh0, wl0, kH / 32, kH, kKT0, kI / 32, n);
    n = (long)kH * kH;  // W_ih1 -> layer1 ktiles [0,128)
    tile_weights<<<(unsigned)((n + kTPB - 1) / kTPB), kTPB, 0, stream>>>(
        W_ih1, wl1, kH / 32, kH, kKT1, 0, n);
    n = (long)kH * kH;  // W_hh1 -> layer1 ktiles [128,256)
    tile_weights<<<(unsigned)((n + kTPB - 1) / kTPB), kTPB, 0, stream>>>(
        W_hh1, wl1, kH / 32, kH, kKT1, kH / 32, n);

    rnn_persistent<<<kNBlk, kTPB, 0, stream>>>(wl0, wl1, z0, z1, bias0, bias1,
                                               h01, xwin, yacc, bar, W_fc, b_fc, out);
}
